// HCDPE_Tokenizer_3204045603805
// MI455X (gfx1250) — matrix operand order repaired and checked by reading, not compiled, not run on hardware
//
#include <hip/hip_runtime.h>
#include <hip/hip_bf16.h>
#include <math.h>

#define BSZ 2048
#define NJ 17
#define TN 34
#define HIDD 512
#define TDD 512
#define KCB 2048
#define TOKI 64
#define NLAYER 4
#define EPSV 1e-5f
#define DECAY_V 0.9f

#define M_ENC (BSZ * NJ)   // 34816
#define M_TOK (BSZ * TN)   // 69632

typedef unsigned short u16;
typedef unsigned int u32;
typedef unsigned long long u64;

typedef __attribute__((ext_vector_type(16))) __bf16 v16bf;
typedef __attribute__((ext_vector_type(8)))  float  v8f;

union FragBF {
  uint4 q[2];
  v16bf v;
};

__device__ __forceinline__ u16 f2bf(float f) {
  u32 u = __float_as_uint(f);
  u += 0x7FFFu + ((u >> 16) & 1u);   // round-to-nearest-even
  return (u16)(u >> 16);
}

__device__ __forceinline__ float gelu_exact(float x) {
  return 0.5f * x * (1.0f + erff(x * 0.70710678118654752f));
}

__device__ __forceinline__ float wave_sum(float v) {
#pragma unroll
  for (int m = 16; m >= 1; m >>= 1) v += __shfl_xor(v, m, 32);
  return v;
}

// ---------------------------------------------------------------------------
// CDNA5 async memory->LDS staging (GLOBAL_LOAD_ASYNC_TO_LDS_B128, ASYNCcnt)
// ---------------------------------------------------------------------------
#define LDS_STRIDE 40   // u16 units -> 80B rows: 16B aligned, conflict-light

__device__ __forceinline__ u32 lds_off32(const u16* p) {
  // flat shared-space pointer: low 32 bits == workgroup-relative LDS offset
  return (u32)(unsigned long long)(uintptr_t)p;
}

__device__ __forceinline__ void async_ld_b128(u32 lds_addr, u64 gaddr) {
  asm volatile("global_load_async_to_lds_b128 %0, %1, off"
               :: "v"(lds_addr), "v"(gaddr) : "memory");
}

__device__ __forceinline__ void wait_async_le4() {
  asm volatile("s_wait_asynccnt 0x4" ::: "memory");
}
__device__ __forceinline__ void wait_async_0() {
  asm volatile("s_wait_asynccnt 0x0" ::: "memory");
}

// stage one 128x32 bf16 tile of A and one of Bt into LDS (4 async ops / wave)
__device__ __forceinline__ void stage_async(u32 ldsA, u32 ldsB,
                                            const u16* __restrict__ A,
                                            const u16* __restrict__ Bt,
                                            int m0, int n0, int Kd, int k0, int tid) {
  for (int i = tid; i < 512; i += 256) {
    int row = i >> 2, seg = i & 3;
    u32 loff = (u32)((row * LDS_STRIDE + seg * 8) * 2);
    u64 ga = (u64)(uintptr_t)(A  + (size_t)(m0 + row) * Kd + k0 + seg * 8);
    u64 gb = (u64)(uintptr_t)(Bt + (size_t)(n0 + row) * Kd + k0 + seg * 8);
    async_ld_b128(ldsA + loff, ga);
    async_ld_b128(ldsB + loff, gb);
  }
}

// ---------------------------------------------------------------------------
// Generic bf16 WMMA GEMM: C[M,N] = op(A[M,K] @ Bt[N,K]^T + bias (+Cres))
// block tile 128x128, 8 waves each own 16 rows x 128 cols (8 C tiles of 16x16)
// double-buffered async LDS staging
// ---------------------------------------------------------------------------
__global__ __launch_bounds__(256)
void gemm_bf16_kernel(const u16* __restrict__ A,     // [M][K] bf16 row-major
                      const u16* __restrict__ Bt,    // [N][K] bf16 (B transposed)
                      const float* __restrict__ bias,// [N]
                      const float* __restrict__ Cres,// [M][N] or null
                      float* __restrict__ Cout,      // [M][N] or null
                      u16* __restrict__ Obf,         // [M][N] bf16 or null
                      int M, int N, int Kd, int do_gelu)
{
  __shared__ __align__(16) u16 sA[2][128 * LDS_STRIDE];
  __shared__ __align__(16) u16 sB[2][128 * LDS_STRIDE];

  const int tid  = threadIdx.x;
  const int lane = tid & 31;
  const int w    = tid >> 5;
  const int half = lane >> 4;
  const int l15  = lane & 15;

  const int n0 = blockIdx.x * 128;
  const int m0 = blockIdx.y * 128;

  const u32 ldsA0 = lds_off32(&sA[0][0]), ldsA1 = lds_off32(&sA[1][0]);
  const u32 ldsB0 = lds_off32(&sB[0][0]), ldsB1 = lds_off32(&sB[1][0]);

  v8f acc[8];
#pragma unroll
  for (int t = 0; t < 8; ++t)
#pragma unroll
    for (int e = 0; e < 8; ++e) acc[t][e] = 0.0f;

  const int NK = Kd >> 5;
  stage_async(ldsA0, ldsB0, A, Bt, m0, n0, Kd, 0, tid);

  for (int kk = 0; kk < NK; ++kk) {
    const int buf = kk & 1;
    const bool has_next = (kk + 1 < NK);
    if (has_next)
      stage_async(buf ? ldsA0 : ldsA1, buf ? ldsB0 : ldsB1,
                  A, Bt, m0, n0, Kd, (kk + 1) << 5, tid);
    if (has_next) wait_async_le4(); else wait_async_0();
    __syncthreads();

    // A fragment (16x32 bf16): lanes 0-15 rows M0..15 K 0..7/16..23,
    // lanes 16-31 same rows K 8..15/24..31
    FragBF fa;
    const u16* pa = &sA[buf][(w * 16 + l15) * LDS_STRIDE + half * 8];
    fa.q[0] = *(const uint4*)(pa);
    fa.q[1] = *(const uint4*)(pa + 16);

#pragma unroll
    for (int t = 0; t < 8; ++t) {
      FragBF fb;
      const u16* pb = &sB[buf][(t * 16 + l15) * LDS_STRIDE + half * 8];
      fb.q[0] = *(const uint4*)(pb);
      fb.q[1] = *(const uint4*)(pb + 16);
      acc[t] = __builtin_amdgcn_wmma_f32_16x16x32_bf16(
          false, fa.v, false, fb.v, (short)0, acc[t], false, false);
    }
    __syncthreads();
  }

  // epilogue: vgpr r of lane -> row = r + (half ? 8 : 0), col = l15 (per tile)
#pragma unroll
  for (int t = 0; t < 8; ++t) {
    int col = n0 + t * 16 + l15;
    float bv = bias ? bias[col] : 0.0f;
#pragma unroll
    for (int r = 0; r < 8; ++r) {
      int row = m0 + w * 16 + r + half * 8;
      float v = acc[t][r] + bv;
      if (Cres) v += Cres[(size_t)row * N + col];
      if (do_gelu) v = gelu_exact(v);
      if (Cout) Cout[(size_t)row * N + col] = v;
      if (Obf)  Obf[(size_t)row * N + col] = f2bf(v);
    }
  }
}

// ---------------------------------------------------------------------------
// VQ nearest-codebook: argmin_k ( ||cb_k||^2 - 2 ef_i . cb_k )
// Same WMMA tiling; running per-row best across 16 codebook chunks.
// ---------------------------------------------------------------------------
__global__ __launch_bounds__(256)
void vq_argmin_kernel(const u16* __restrict__ efb,   // [M][TD] bf16
                      const u16* __restrict__ cbb,   // [K][TD] bf16
                      const float* __restrict__ cbn, // [K] ||cb||^2
                      int* __restrict__ idx_out,
                      float* __restrict__ idxf_out)
{
  __shared__ __align__(16) u16 sA[2][128 * LDS_STRIDE];
  __shared__ __align__(16) u16 sB[2][128 * LDS_STRIDE];

  const int tid  = threadIdx.x;
  const int lane = tid & 31;
  const int w    = tid >> 5;
  const int half = lane >> 4;
  const int l15  = lane & 15;
  const int m0   = blockIdx.x * 128;

  const u32 ldsA0 = lds_off32(&sA[0][0]), ldsA1 = lds_off32(&sA[1][0]);
  const u32 ldsB0 = lds_off32(&sB[0][0]), ldsB1 = lds_off32(&sB[1][0]);

  float bestv[8];
  int   bestk[8];
#pragma unroll
  for (int r = 0; r < 8; ++r) { bestv[r] = 3.4e38f; bestk[r] = 0; }

  for (int chunk = 0; chunk < KCB / 128; ++chunk) {
    const int n0 = chunk * 128;
    v8f acc[8];
#pragma unroll
    for (int t = 0; t < 8; ++t)
#pragma unroll
      for (int e = 0; e < 8; ++e) acc[t][e] = 0.0f;

    const int NK = TDD >> 5;
    stage_async(ldsA0, ldsB0, efb, cbb, m0, n0, TDD, 0, tid);

    for (int kk = 0; kk < NK; ++kk) {
      const int buf = kk & 1;
      const bool has_next = (kk + 1 < NK);
      if (has_next)
        stage_async(buf ? ldsA0 : ldsA1, buf ? ldsB0 : ldsB1,
                    efb, cbb, m0, n0, TDD, (kk + 1) << 5, tid);
      if (has_next) wait_async_le4(); else wait_async_0();
      __syncthreads();

      FragBF fa;
      const u16* pa = &sA[buf][(w * 16 + l15) * LDS_STRIDE + half * 8];
      fa.q[0] = *(const uint4*)(pa);
      fa.q[1] = *(const uint4*)(pa + 16);
#pragma unroll
      for (int t = 0; t < 8; ++t) {
        FragBF fb;
        const u16* pb = &sB[buf][(t * 16 + l15) * LDS_STRIDE + half * 8];
        fb.q[0] = *(const uint4*)(pb);
        fb.q[1] = *(const uint4*)(pb + 16);
        acc[t] = __builtin_amdgcn_wmma_f32_16x16x32_bf16(
            false, fa.v, false, fb.v, (short)0, acc[t], false, false);
      }
      __syncthreads();
    }

#pragma unroll
    for (int r = 0; r < 8; ++r) {
      float bv = 3.4e38f; int bk = 0;
#pragma unroll
      for (int t = 0; t < 8; ++t) {
        int col = n0 + t * 16 + l15;
        float dist = cbn[col] - 2.0f * acc[t][r];
        if (dist < bv || (dist == bv && col < bk)) { bv = dist; bk = col; }
      }
      // reduce across the 16 lanes of this half (they hold the 16 columns)
#pragma unroll
      for (int m = 1; m <= 8; m <<= 1) {
        float ov = __shfl_xor(bv, m, 32);
        int   ok = __shfl_xor(bk, m, 32);
        if (ov < bv || (ov == bv && ok < bk)) { bv = ov; bk = ok; }
      }
      if (bv < bestv[r]) { bestv[r] = bv; bestk[r] = bk; }
    }
  }

  if (l15 == 0) {
#pragma unroll
    for (int r = 0; r < 8; ++r) {
      int row = m0 + w * 16 + r + half * 8;
      idx_out[row]  = bestk[r];
      idxf_out[row] = (float)bestk[r];
    }
  }
}

// ---------------------------------------------------------------------------
// LayerNorm (one wave per 512-ch row), optional residual add + bf16 pack
// ---------------------------------------------------------------------------
__global__ __launch_bounds__(256)
void ln_kernel(const float* __restrict__ X, const float* __restrict__ Yadd,
               float* __restrict__ xpy_out, float* __restrict__ f32_out,
               u16* __restrict__ bf_out,
               const float* __restrict__ g, const float* __restrict__ b, int rows)
{
  int wid = blockIdx.x * (blockDim.x >> 5) + (threadIdx.x >> 5);
  if (wid >= rows) return;
  int lane = threadIdx.x & 31;
  const size_t base = (size_t)wid * HIDD;
  float v[16];
  float s = 0.f, sq = 0.f;
#pragma unroll
  for (int i = 0; i < 16; ++i) {
    int c = lane + i * 32;
    float x = X[base + c];
    if (Yadd) x += Yadd[base + c];
    v[i] = x; s += x; sq += x * x;
  }
  if (xpy_out) {
#pragma unroll
    for (int i = 0; i < 16; ++i) xpy_out[base + lane + i * 32] = v[i];
  }
  s = wave_sum(s); sq = wave_sum(sq);
  float mean = s * (1.0f / HIDD);
  float var  = sq * (1.0f / HIDD) - mean * mean;
  float rs   = rsqrtf(var + EPSV);
#pragma unroll
  for (int i = 0; i < 16; ++i) {
    int c = lane + i * 32;
    float y = (v[i] - mean) * rs * g[c] + b[c];
    if (f32_out) f32_out[base + c] = y;
    if (bf_out)  bf_out[base + c]  = f2bf(y);
  }
}

// token mixing MLP over joints: (17 -> 64 -> 17) per (batch, channel), in-place
__global__ __launch_bounds__(256)
void tokenmix_kernel(float* __restrict__ Y,
                     const float* __restrict__ w1, const float* __restrict__ b1,
                     const float* __restrict__ w2, const float* __restrict__ b2)
{
  __shared__ float sw1[NJ * TOKI], sb1[TOKI], sw2[TOKI * NJ], sb2[NJ];
  int tid = threadIdx.x;
  for (int i = tid; i < NJ * TOKI; i += 256) { sw1[i] = w1[i]; sw2[i] = w2[i]; }
  if (tid < TOKI) sb1[tid] = b1[tid];
  if (tid < NJ)   sb2[tid] = b2[tid];
  __syncthreads();
  int gid = blockIdx.x * 256 + tid;
  int b = gid >> 9, c = gid & 511;
  float in[NJ];
  for (int j = 0; j < NJ; ++j) in[j] = Y[((size_t)b * NJ + j) * HIDD + c];
  float h[TOKI];
  for (int u = 0; u < TOKI; ++u) {
    float a = sb1[u];
#pragma unroll
    for (int j = 0; j < NJ; ++j) a += in[j] * sw1[j * TOKI + u];
    h[u] = gelu_exact(a);
  }
  for (int j = 0; j < NJ; ++j) {
    float a = sb2[j];
#pragma unroll
    for (int u = 0; u < TOKI; ++u) a += h[u] * sw2[u * NJ + j];
    Y[((size_t)b * NJ + j) * HIDD + c] = a;
  }
}

// token expand 17 -> 34 per (batch, channel), writes bf16 for feat GEMM
__global__ __launch_bounds__(256)
void tokexpand_kernel(const float* __restrict__ X,
                      const float* __restrict__ tw, const float* __restrict__ tb,
                      u16* __restrict__ Obf)
{
  __shared__ float sw[NJ * TN], sb[TN];
  int tid = threadIdx.x;
  for (int i = tid; i < NJ * TN; i += 256) sw[i] = tw[i];
  if (tid < TN) sb[tid] = tb[tid];
  __syncthreads();
  int gid = blockIdx.x * 256 + tid;
  int b = gid >> 9, c = gid & 511;
  float in[NJ];
  for (int j = 0; j < NJ; ++j) in[j] = X[((size_t)b * NJ + j) * HIDD + c];
  for (int t = 0; t < TN; ++t) {
    float a = sb[t];
#pragma unroll
    for (int j = 0; j < NJ; ++j) a += in[j] * sw[j * TN + t];
    Obf[((size_t)b * TN + t) * HIDD + c] = f2bf(a);
  }
}

// decoder token map 34 -> 17 on quantized codes (gather), bf16 out for decstart
__global__ __launch_bounds__(256)
void dectok_kernel(const float* __restrict__ cb, const int* __restrict__ idx,
                   const float* __restrict__ dwt, const float* __restrict__ db,
                   u16* __restrict__ Obf)
{
  __shared__ float sw[TN * NJ], sb[NJ];
  int tid = threadIdx.x;
  for (int i = tid; i < TN * NJ; i += 256) sw[i] = dwt[i];
  if (tid < NJ) sb[tid] = db[tid];
  __syncthreads();
  int gid = blockIdx.x * 256 + tid;
  int b = gid >> 9, c = gid & 511;
  float in[TN];
  for (int t = 0; t < TN; ++t)
    in[t] = cb[(size_t)idx[b * TN + t] * TDD + c];
  for (int j = 0; j < NJ; ++j) {
    float a = sb[j];
#pragma unroll
    for (int t = 0; t < TN; ++t) a += in[t] * sw[t * NJ + j];
    Obf[((size_t)b * NJ + j) * HIDD + c] = f2bf(a);
  }
}

__global__ void embed_kernel(const float* __restrict__ joints,
                             const float* __restrict__ invtok,
                             const float* __restrict__ sw, const float* __restrict__ sb,
                             float* __restrict__ X)
{
  int gid = blockIdx.x * blockDim.x + threadIdx.x;
  int row = gid >> 9, c = gid & 511;
  float c0  = joints[row * 3 + 0];
  float c1  = joints[row * 3 + 1];
  float vis = joints[row * 3 + 2];
  float v = c0 * sw[c] + c1 * sw[HIDD + c] + sb[c];
  if (!(vis > 0.0f)) v = invtok[c];
  X[(size_t)gid] = v;
}

__global__ void pack_bf16_kernel(const float* __restrict__ src, u16* __restrict__ dst, int n) {
  int i = blockIdx.x * blockDim.x + threadIdx.x;
  if (i < n) dst[i] = f2bf(src[i]);
}

__global__ void wtranspose_kernel(const float* __restrict__ W, u16* __restrict__ WT,
                                  int Kd, int Nd) {
  int i = blockIdx.x * blockDim.x + threadIdx.x;
  if (i >= Kd * Nd) return;
  int k = i / Nd, n = i - k * Nd;
  WT[(size_t)n * Kd + k] = f2bf(W[i]);
}

__global__ void cbnorm_kernel(const float* __restrict__ cb, float* __restrict__ cbn) {
  int wid = blockIdx.x * 8 + (threadIdx.x >> 5);
  int lane = threadIdx.x & 31;
  float s = 0.f;
#pragma unroll
  for (int i = 0; i < 16; ++i) {
    float v = cb[(size_t)wid * TDD + lane + i * 32];
    s += v * v;
  }
  s = wave_sum(s);
  if (lane == 0) cbn[wid] = s;
}

// per-row segment sums: counts[k] += 1, dw[k,:] += ef[i,:], loss accum
__global__ __launch_bounds__(256)
void stats_kernel(const float* __restrict__ ef, const float* __restrict__ cb,
                  const int* __restrict__ idx,
                  float* __restrict__ counts, float* __restrict__ dw,
                  float* __restrict__ loss_acc)
{
  int wid = blockIdx.x * 8 + (threadIdx.x >> 5);
  int lane = threadIdx.x & 31;
  int k = idx[wid];
  float s = 0.f;
#pragma unroll
  for (int i = 0; i < 16; ++i) {
    int c = lane + i * 32;
    float e = ef[(size_t)wid * TDD + c];
    atomicAdd(&dw[(size_t)k * TDD + c], e);
    float d = cb[(size_t)k * TDD + c] - e;
    s += d * d;
  }
  s = wave_sum(s);
  if (lane == 0) { atomicAdd(loss_acc, s); atomicAdd(&counts[k], 1.0f); }
}

__global__ void ema_pre_kernel(const float* __restrict__ ema_cs, float* __restrict__ counts,
                               float* __restrict__ n_acc)
{
  int k = blockIdx.x * blockDim.x + threadIdx.x;
  float raw = ema_cs[k] * DECAY_V + (1.0f - DECAY_V) * counts[k];
  counts[k] = raw;
  float s = wave_sum(raw);
  if ((threadIdx.x & 31) == 0) atomicAdd(n_acc, s);
}

__global__ void ema_fin_kernel(const float* __restrict__ ema_w, const float* __restrict__ dwacc,
                               const float* __restrict__ counts, const float* __restrict__ n_acc,
                               const float* __restrict__ loss_acc,
                               float* __restrict__ out_cb, float* __restrict__ out_cs,
                               float* __restrict__ out_ema, float* __restrict__ out_loss)
{
  int t = blockIdx.x * blockDim.x + threadIdx.x;
  int k = t >> 9;
  float n = *n_acc;
  float ncs = (counts[k] + 1e-5f) / (n + (float)KCB * 1e-5f) * n;
  float nw = ema_w[t] * DECAY_V + (1.0f - DECAY_V) * dwacc[t];
  out_ema[t] = nw;
  out_cb[t]  = nw / ncs;
  if ((t & 511) == 0) out_cs[k] = ncs;
  if (t == 0) out_loss[0] = loss_acc[0] / (float)((size_t)M_TOK * TDD);
}

// final LN + 512->2 projection (one wave per row)
__global__ __launch_bounds__(256)
void rec_kernel(const float* __restrict__ Y, const float* __restrict__ g,
                const float* __restrict__ b, const float* __restrict__ rw,
                const float* __restrict__ rb, float* __restrict__ out)
{
  int wid = blockIdx.x * 8 + (threadIdx.x >> 5);
  int lane = threadIdx.x & 31;
  const size_t base = (size_t)wid * HIDD;
  float v[16]; float s = 0.f, sq = 0.f;
#pragma unroll
  for (int i = 0; i < 16; ++i) {
    int c = lane + i * 32;
    float x = Y[base + c];
    v[i] = x; s += x; sq += x * x;
  }
  s = wave_sum(s); sq = wave_sum(sq);
  float mean = s * (1.0f / HIDD);
  float var  = sq * (1.0f / HIDD) - mean * mean;
  float rs   = rsqrtf(var + EPSV);
  float r0 = 0.f, r1 = 0.f;
#pragma unroll
  for (int i = 0; i < 16; ++i) {
    int c = lane + i * 32;
    float y = (v[i] - mean) * rs * g[c] + b[c];
    r0 += y * rw[c * 2 + 0];
    r1 += y * rw[c * 2 + 1];
  }
  r0 = wave_sum(r0); r1 = wave_sum(r1);
  if (lane == 0) {
    out[(size_t)wid * 2 + 0] = r0 + rb[0];
    out[(size_t)wid * 2 + 1] = r1 + rb[1];
  }
}

// ---------------------------------------------------------------------------
extern "C" void kernel_launch(void* const* d_in, const int* in_sizes, int n_in,
                              void* d_out, int out_size, void* d_ws, size_t ws_size,
                              hipStream_t stream) {
  (void)in_sizes; (void)out_size; (void)ws_size;
  const float* in[48];
  for (int i = 0; i < n_in && i < 48; ++i) in[i] = (const float*)d_in[i];
  // input order (nested dicts flattened in insertion order):
  // 0 joints 1 joints_feature 2 cls_logits 3 invisible_token 4 start_w 5 start_b
  // 6..17 enc_params{ln1_g,ln1_b,tok_w1,tok_b1,tok_w2,tok_b2,ln2_g,ln2_b,ch_w1,ch_b1,ch_w2,ch_b2}
  // 18 enc_ln_g 19 enc_ln_b 20 tokmlp_w 21 tokmlp_b 22 feat_w 23 feat_b
  // 24 codebook 25 ema_cluster_size 26 ema_w 27 dec_tok_w 28 dec_tok_b
  // 29 decstart_w 30 decstart_b 31..42 dec_params 43 dec_ln_g 44 dec_ln_b 45 rec_w 46 rec_b

  char* ws = (char*)d_ws;
  size_t off = 0;
  auto alloc = [&](size_t bytes) { size_t o = off; off += (bytes + 255) & ~(size_t)255; return o; };
  float* bufA  = (float*)(ws + alloc((size_t)M_TOK * TDD * 4)); // f32 activations / ef
  float* bufY  = (float*)(ws + alloc((size_t)M_ENC * HIDD * 4)); // token-mix branch
  u16*   Abf   = (u16*)  (ws + alloc((size_t)M_TOK * TDD * 2)); // bf16 GEMM A
  u16*   Hbf   = (u16*)  (ws + alloc((size_t)M_TOK * TDD * 2)); // bf16 hidden / ef
  u16*   WT1   = (u16*)  (ws + alloc((size_t)HIDD * HIDD * 2));
  u16*   WT2   = (u16*)  (ws + alloc((size_t)HIDD * HIDD * 2));
  u16*   CBbf  = (u16*)  (ws + alloc((size_t)KCB * TDD * 2));
  float* cbn   = (float*)(ws + alloc((size_t)KCB * 4));
  int*   idxb  = (int*)  (ws + alloc((size_t)M_TOK * 4));
  float* counts= (float*)(ws + alloc((size_t)KCB * 4));
  float* dw    = (float*)(ws + alloc((size_t)KCB * TDD * 4));
  float* scal  = (float*)(ws + alloc(64));
  float* loss_acc = scal;
  float* n_acc    = scal + 1;

  float* out = (float*)d_out;
  float* out_rec  = out;                          // 69632
  float* out_idx  = out + 69632;                  // 69632 (idx as float)
  float* out_loss = out + 139264;                 // 1
  float* out_cb   = out + 139265;                 // 1048576
  float* out_cs   = out + 139265 + 1048576;       // 2048
  float* out_ema  = out + 139265 + 1048576 + 2048;// 1048576

  hipMemsetAsync(counts, 0, (size_t)KCB * 4, stream);
  hipMemsetAsync(dw, 0, (size_t)KCB * TDD * 4, stream);
  hipMemsetAsync(scal, 0, 8, stream);

  auto mixer = [&](const float* ln1g, const float* ln1b,
                   const float* tw1, const float* tb1,
                   const float* tw2, const float* tb2,
                   const float* ln2g, const float* ln2b,
                   const float* cw1, const float* cb1,
                   const float* cw2, const float* cb2) {
    ln_kernel<<<4352, 256, 0, stream>>>(bufA, nullptr, nullptr, bufY, nullptr, ln1g, ln1b, M_ENC);
    tokenmix_kernel<<<4096, 256, 0, stream>>>(bufY, tw1, tb1, tw2, tb2);
    // bufA := x + y ; Abf := bf16(ln2(x+y))
    ln_kernel<<<4352, 256, 0, stream>>>(bufA, bufY, bufA, nullptr, Abf, ln2g, ln2b, M_ENC);
    wtranspose_kernel<<<1024, 256, 0, stream>>>(cw1, WT1, HIDD, HIDD);
    wtranspose_kernel<<<1024, 256, 0, stream>>>(cw2, WT2, HIDD, HIDD);
    gemm_bf16_kernel<<<dim3(4, M_ENC / 128), 256, 0, stream>>>(
        Abf, WT1, cb1, nullptr, nullptr, Hbf, M_ENC, HIDD, HIDD, 1);
    gemm_bf16_kernel<<<dim3(4, M_ENC / 128), 256, 0, stream>>>(
        Hbf, WT2, cb2, bufA, bufA, nullptr, M_ENC, HIDD, HIDD, 0);
  };

  // ---- encoder ----
  embed_kernel<<<(M_ENC * HIDD) / 256, 256, 0, stream>>>(in[0], in[3], in[4], in[5], bufA);
  for (int l = 0; l < NLAYER; ++l)
    mixer(in[6] + l * HIDD, in[7] + l * HIDD,
          in[8] + l * NJ * TOKI, in[9] + l * TOKI,
          in[10] + l * TOKI * NJ, in[11] + l * NJ,
          in[12] + l * HIDD, in[13] + l * HIDD,
          in[14] + (size_t)l * HIDD * HIDD, in[15] + l * HIDD,
          in[16] + (size_t)l * HIDD * HIDD, in[17] + l * HIDD);
  ln_kernel<<<4352, 256, 0, stream>>>(bufA, nullptr, nullptr, bufA, nullptr, in[18], in[19], M_ENC);
  tokexpand_kernel<<<4096, 256, 0, stream>>>(bufA, in[20], in[21], Abf);

  // ---- feat + VQ ----
  wtranspose_kernel<<<1024, 256, 0, stream>>>(in[22], WT1, HIDD, TDD);
  gemm_bf16_kernel<<<dim3(4, M_TOK / 128), 256, 0, stream>>>(
      Abf, WT1, in[23], nullptr, bufA, Hbf, M_TOK, TDD, HIDD, 0);
  pack_bf16_kernel<<<(KCB * TDD) / 256, 256, 0, stream>>>(in[24], CBbf, KCB * TDD);
  cbnorm_kernel<<<KCB / 8, 256, 0, stream>>>(in[24], cbn);
  vq_argmin_kernel<<<M_TOK / 128, 256, 0, stream>>>(Hbf, CBbf, cbn, idxb, out_idx);
  stats_kernel<<<M_TOK / 8, 256, 0, stream>>>(bufA, in[24], idxb, counts, dw, loss_acc);

  // ---- decoder ----
  dectok_kernel<<<4096, 256, 0, stream>>>(in[24], idxb, in[27], in[28], Abf);
  wtranspose_kernel<<<1024, 256, 0, stream>>>(in[29], WT1, TDD, HIDD);
  gemm_bf16_kernel<<<dim3(4, M_ENC / 128), 256, 0, stream>>>(
      Abf, WT1, in[30], nullptr, bufA, nullptr, M_ENC, HIDD, TDD, 0);
  for (int l = 0; l < NLAYER; ++l)
    mixer(in[31] + l * HIDD, in[32] + l * HIDD,
          in[33] + l * NJ * TOKI, in[34] + l * TOKI,
          in[35] + l * TOKI * NJ, in[36] + l * NJ,
          in[37] + l * HIDD, in[38] + l * HIDD,
          in[39] + (size_t)l * HIDD * HIDD, in[40] + l * HIDD,
          in[41] + (size_t)l * HIDD * HIDD, in[42] + l * HIDD);
  rec_kernel<<<4352, 256, 0, stream>>>(bufA, in[43], in[44], in[45], in[46], out_rec);

  // ---- EMA / loss outputs ----
  ema_pre_kernel<<<KCB / 256, 256, 0, stream>>>(in[25], counts, n_acc);
  ema_fin_kernel<<<(KCB * TDD) / 256, 256, 0, stream>>>(
      in[26], dw, counts, n_acc, loss_acc, out_cb, out_cs, out_ema, out_loss);
}
